// LinearAttention_86517821215774
// MI455X (gfx1250) — compile-verified, matrix operand order checked by reading, not run on hardware
//
#include <hip/hip_runtime.h>

// ---------------------------------------------------------------------------
// LinearAttention on gfx1250 (MI455X): bf16 WMMA pipeline
//   qkv = x @ Wqkv                      (WMMA GEMM, double-buffered LDS)
//   q   = softmax_d(q * 1/8)            (wave32 shuffle softmax)
//   k   = softmax_n(k), masked          (block column softmax)
//   ctx = q^T @ k  per (b,h)            (WMMA, deterministic split-K)
//   att = v @ ctx  per (b,h)            (WMMA)
//   out = att @ Wout + bout             (WMMA GEMM + bias, double-buffered)
// ---------------------------------------------------------------------------

typedef __attribute__((ext_vector_type(16))) __bf16 v16bf;
typedef __attribute__((ext_vector_type(8)))  float  v8f;

union FragAB { v16bf v; unsigned u[8]; };
union FragC  { v8f v;  float f[8]; };

__device__ inline unsigned short f2bf(float x) {
  unsigned u = __float_as_uint(x);
  u += 0x7FFFu + ((u >> 16) & 1u);         // round-to-nearest-even
  return (unsigned short)(u >> 16);
}

__device__ inline void zero_frag(FragC& c) {
#pragma unroll
  for (int i = 0; i < 8; ++i) c.f[i] = 0.0f;
}

// ---------------------------------------------------------------------------
// Generic GEMM: C[M,N] = A[M,K] @ B[K,N] (+bias), A/B row-major f32,
// converted to bf16 into double-buffered LDS. 256 threads = 8 waves;
// block tile 128x128x32; wave (wm,wn) in 2x4 grid computes 64x32 output.
// One __syncthreads per K-step; global loads for step i+1 overlap WMMA of i.
// ---------------------------------------------------------------------------
template <bool HAS_BIAS>
__global__ __launch_bounds__(256)
void gemm_bf16_wmma(const float* __restrict__ A, const float* __restrict__ B,
                    const float* __restrict__ bias, float* __restrict__ C,
                    int M, int N, int K) {
  __shared__ unsigned short As[2][128][34];
  __shared__ unsigned short Bs[2][128][34];
  const int t = threadIdx.x, lane = t & 31, w = t >> 5;
  const int wm = w >> 2, wn = w & 3;
  const int m0 = blockIdx.x * 128, n0 = blockIdx.y * 128;

  FragC acc[4][2];
#pragma unroll
  for (int mt = 0; mt < 4; ++mt) { zero_frag(acc[mt][0]); zero_frag(acc[mt][1]); }

  const int ar = t >> 3, ac = (t & 7) << 2;     // A: 128 rows x 8 float4
  const int br = t >> 5, bc = (t & 31) << 2;    // B: 32 rows x 32 float4
  const int kd = (lane >> 4) << 2;              // lane -> dword offset in row
  const int nk = K >> 5;

  float4 a4[4], b4[4];                          // staging registers

  auto gload = [&](int k0) {
#pragma unroll
    for (int i = 0; i < 4; ++i)
      a4[i] = *(const float4*)(A + (size_t)(m0 + ar + (i << 5)) * K + k0 + ac);
#pragma unroll
    for (int i = 0; i < 4; ++i)
      b4[i] = *(const float4*)(B + (size_t)(k0 + br + (i << 3)) * N + n0 + bc);
  };
  auto lstore = [&](int buf) {
#pragma unroll
    for (int i = 0; i < 4; ++i) {
      const int row = ar + (i << 5);
      As[buf][row][ac + 0] = f2bf(a4[i].x); As[buf][row][ac + 1] = f2bf(a4[i].y);
      As[buf][row][ac + 2] = f2bf(a4[i].z); As[buf][row][ac + 3] = f2bf(a4[i].w);
    }
#pragma unroll
    for (int i = 0; i < 4; ++i) {
      const int kk = br + (i << 3);
      Bs[buf][bc + 0][kk] = f2bf(b4[i].x); Bs[buf][bc + 1][kk] = f2bf(b4[i].y);
      Bs[buf][bc + 2][kk] = f2bf(b4[i].z); Bs[buf][bc + 3][kk] = f2bf(b4[i].w);
    }
  };

  gload(0);
  lstore(0);
  __syncthreads();

  for (int it = 0; it < nk; ++it) {
    const int p = it & 1;
    if (it + 1 < nk) gload((it + 1) << 5);      // overlap with WMMA below
    if (it + 2 < nk) {                          // global_prefetch_b8 two ahead
      __builtin_prefetch(A + (size_t)(m0 + ar) * K + ((it + 2) << 5) + ac, 0, 1);
      __builtin_prefetch(B + (size_t)(((it + 2) << 5) + br) * N + n0 + bc, 0, 1);
    }

    FragAB bf0, bf1;
    {
      const unsigned* p0 = (const unsigned*)&Bs[p][wn * 32 + (lane & 15)][0];
      const unsigned* p1 = (const unsigned*)&Bs[p][wn * 32 + 16 + (lane & 15)][0];
#pragma unroll
      for (int j = 0; j < 4; ++j) {
        bf0.u[j] = p0[kd + j]; bf0.u[4 + j] = p0[8 + kd + j];
        bf1.u[j] = p1[kd + j]; bf1.u[4 + j] = p1[8 + kd + j];
      }
    }
#pragma unroll
    for (int mt = 0; mt < 4; ++mt) {
      FragAB af;
      const unsigned* pa = (const unsigned*)&As[p][wm * 64 + mt * 16 + (lane & 15)][0];
#pragma unroll
      for (int j = 0; j < 4; ++j) { af.u[j] = pa[kd + j]; af.u[4 + j] = pa[8 + kd + j]; }
      acc[mt][0].v = __builtin_amdgcn_wmma_f32_16x16x32_bf16(
          false, af.v, false, bf0.v, (short)0, acc[mt][0].v, false, false);
      acc[mt][1].v = __builtin_amdgcn_wmma_f32_16x16x32_bf16(
          false, af.v, false, bf1.v, (short)0, acc[mt][1].v, false, false);
    }

    if (it + 1 < nk) lstore(1 - p);             // buffer 1-p idle this step
    __syncthreads();
  }

  const int chi = (lane >> 4) << 3;
#pragma unroll
  for (int mt = 0; mt < 4; ++mt)
#pragma unroll
    for (int nt = 0; nt < 2; ++nt) {
      const int col = n0 + wn * 32 + nt * 16 + (lane & 15);
      float badd = 0.0f;
      if (HAS_BIAS) badd = bias[col];
#pragma unroll
      for (int r = 0; r < 8; ++r) {
        const int row = m0 + wm * 64 + mt * 16 + chi + r;
        C[(size_t)row * N + col] = acc[mt][nt].f[r] + badd;
      }
    }
}

// ---------------------------------------------------------------------------
// q softmax over feature dim (64), scaled by 1/8. One wave per row.
// ---------------------------------------------------------------------------
__global__ __launch_bounds__(256)
void q_softmax_kernel(float* __restrict__ qkv) {
  const int t = threadIdx.x, lane = t & 31, w = t >> 5;
  const int rowid = blockIdx.x * 8 + w;              // 0..262143 = (b*N+n)*8+h
  const int bn = rowid >> 3, h = rowid & 7;
  float* p = qkv + (size_t)bn * 1536 + h * 64;
  float x0 = p[lane] * 0.125f;
  float x1 = p[lane + 32] * 0.125f;
  float m = fmaxf(x0, x1);
#pragma unroll
  for (int off = 16; off > 0; off >>= 1) m = fmaxf(m, __shfl_xor(m, off, 32));
  const float e0 = __expf(x0 - m), e1 = __expf(x1 - m);
  float s = e0 + e1;
#pragma unroll
  for (int off = 16; off > 0; off >>= 1) s += __shfl_xor(s, off, 32);
  const float inv = 1.0f / s;
  p[lane] = e0 * inv;
  p[lane + 32] = e1 * inv;
}

// ---------------------------------------------------------------------------
// k softmax over sequence dim (8192) + mask zeroing. One block per (b,h,e).
// ---------------------------------------------------------------------------
__global__ __launch_bounds__(256)
void k_softmax_kernel(float* __restrict__ qkv, const unsigned char* __restrict__ mask) {
  __shared__ float red[256];
  const int t = threadIdx.x;
  const int c = blockIdx.x;                 // 0..2047
  const int b = c >> 9, he = c & 511, h = he >> 6;
  float* base = qkv + (size_t)b * 8192 * 1536 + 512 + he;
  const unsigned char* mrow = mask + (size_t)(b * 8 + h) * 8192;

  float vals[32];
  float mx = -3.4e38f;
#pragma unroll
  for (int i = 0; i < 32; ++i) {
    vals[i] = base[(size_t)(i * 256 + t) * 1536];
    mx = fmaxf(mx, vals[i]);
  }
  red[t] = mx; __syncthreads();
  for (int s = 128; s > 0; s >>= 1) {
    if (t < s) red[t] = fmaxf(red[t], red[t + s]);
    __syncthreads();
  }
  mx = red[0]; __syncthreads();

  float sum = 0.0f;
#pragma unroll
  for (int i = 0; i < 32; ++i) { vals[i] = __expf(vals[i] - mx); sum += vals[i]; }
  red[t] = sum; __syncthreads();
  for (int s = 128; s > 0; s >>= 1) {
    if (t < s) red[t] += red[t + s];
    __syncthreads();
  }
  const float inv = 1.0f / red[0];
#pragma unroll
  for (int i = 0; i < 32; ++i) {
    const int n = i * 256 + t;
    base[(size_t)n * 1536] = mrow[n] ? 0.0f : vals[i] * inv;
  }
}

// ---------------------------------------------------------------------------
// ctx_partial[p][b,h][d,e] = sum over 1024 n of q[n,d]*k[n,e].
// grid (32 bh, 8 split-K). q/k tiles transposed into LDS as [feat][n].
// Wave w: d-tile = w>>1, e-tiles = 2*(w&1), 2*(w&1)+1.
// ---------------------------------------------------------------------------
__global__ __launch_bounds__(256)
void ctx_partial_kernel(const float* __restrict__ qkv, float* __restrict__ ctxp) {
  __shared__ unsigned short Qs[64][34];
  __shared__ unsigned short Ks[64][34];
  const int t = threadIdx.x, lane = t & 31, w = t >> 5;
  const int bh = blockIdx.x;
  const int b = bh >> 3, h = bh & 7;
  const int nbase = blockIdx.y * 1024;
  const int qcol = h * 64, kcol = 512 + h * 64;
  const int dt = w >> 1, e0 = (w & 1) * 2;
  const int dl = t & 63, nb = t >> 6;
  const int kd = (lane >> 4) << 2;

  FragC acc0, acc1;
  zero_frag(acc0); zero_frag(acc1);

  for (int n0 = 0; n0 < 1024; n0 += 32) {
#pragma unroll
    for (int i = 0; i < 8; ++i) {
      const int n = nb * 8 + i;
      const size_t row = (size_t)(b * 8192 + nbase + n0 + n) * 1536;
      Qs[dl][n] = f2bf(qkv[row + qcol + dl]);
      Ks[dl][n] = f2bf(qkv[row + kcol + dl]);
    }
    __syncthreads();

    FragAB af, bf0, bf1;
    const unsigned* pa = (const unsigned*)&Qs[dt * 16 + (lane & 15)][0];
    const unsigned* p0 = (const unsigned*)&Ks[e0 * 16 + (lane & 15)][0];
    const unsigned* p1 = (const unsigned*)&Ks[e0 * 16 + 16 + (lane & 15)][0];
#pragma unroll
    for (int j = 0; j < 4; ++j) {
      af.u[j]  = pa[kd + j]; af.u[4 + j]  = pa[8 + kd + j];
      bf0.u[j] = p0[kd + j]; bf0.u[4 + j] = p0[8 + kd + j];
      bf1.u[j] = p1[kd + j]; bf1.u[4 + j] = p1[8 + kd + j];
    }
    acc0.v = __builtin_amdgcn_wmma_f32_16x16x32_bf16(
        false, af.v, false, bf0.v, (short)0, acc0.v, false, false);
    acc1.v = __builtin_amdgcn_wmma_f32_16x16x32_bf16(
        false, af.v, false, bf1.v, (short)0, acc1.v, false, false);
    __syncthreads();
  }

  const int chi = (lane >> 4) << 3;
  float* dst = ctxp + ((size_t)blockIdx.y * 32 + bh) * 4096;
#pragma unroll
  for (int r = 0; r < 8; ++r) {
    const int drow = dt * 16 + chi + r;
    dst[drow * 64 + e0 * 16 + (lane & 15)]        = acc0.f[r];
    dst[drow * 64 + (e0 + 1) * 16 + (lane & 15)]  = acc1.f[r];
  }
}

__global__ __launch_bounds__(256)
void ctx_reduce_kernel(const float* __restrict__ ctxp, float* __restrict__ ctx) {
  const int i = blockIdx.x * 256 + threadIdx.x;    // 131072 elements
  float s = 0.0f;
#pragma unroll
  for (int p = 0; p < 8; ++p) s += ctxp[(size_t)p * 131072 + i];
  ctx[i] = s;
}

// ---------------------------------------------------------------------------
// att[n,e] = sum_d v[n,d] * ctx[d,e]   per (b,h). K=64 entirely in LDS.
// grid (64 m-blocks, 32 bh); wave w owns 16-row strip, 4 e-tiles.
// Writes into [b, n, h*64+e] layout for the final GEMM.
// ---------------------------------------------------------------------------
__global__ __launch_bounds__(256)
void attn_out_kernel(const float* __restrict__ qkv, const float* __restrict__ ctx,
                     float* __restrict__ attn) {
  __shared__ unsigned short Vs[128][66];   // [m][d]
  __shared__ unsigned short Cs[64][66];    // [e][d]  (ctx transposed)
  const int t = threadIdx.x, lane = t & 31, w = t >> 5;
  const int mblk = blockIdx.x, bh = blockIdx.y;
  const int b = bh >> 3, h = bh & 7;
  const int vcol = 1024 + h * 64;
  const int d = t & 63, g4 = t >> 6;

#pragma unroll
  for (int i = 0; i < 32; ++i) {
    const int m = g4 * 32 + i;
    const size_t row = (size_t)(b * 8192 + mblk * 128 + m) * 1536;
    Vs[m][d] = f2bf(qkv[row + vcol + d]);
  }
  const float* cbase = ctx + (size_t)bh * 4096;
#pragma unroll
  for (int i = 0; i < 16; ++i) {
    const int e = g4 * 16 + i;
    Cs[e][d] = f2bf(cbase[d * 64 + e]);
  }
  __syncthreads();

  FragC acc[4];
#pragma unroll
  for (int et = 0; et < 4; ++et) zero_frag(acc[et]);

#pragma unroll
  for (int kk = 0; kk < 64; kk += 32) {
    const int kd = (kk >> 1) + ((lane >> 4) << 2);
    FragAB af;
    const unsigned* pa = (const unsigned*)&Vs[w * 16 + (lane & 15)][0];
#pragma unroll
    for (int j = 0; j < 4; ++j) { af.u[j] = pa[kd + j]; af.u[4 + j] = pa[8 + kd + j]; }
#pragma unroll
    for (int et = 0; et < 4; ++et) {
      FragAB bf;
      const unsigned* pb = (const unsigned*)&Cs[et * 16 + (lane & 15)][0];
#pragma unroll
      for (int j = 0; j < 4; ++j) { bf.u[j] = pb[kd + j]; bf.u[4 + j] = pb[8 + kd + j]; }
      acc[et].v = __builtin_amdgcn_wmma_f32_16x16x32_bf16(
          false, af.v, false, bf.v, (short)0, acc[et].v, false, false);
    }
  }

  const int chi = (lane >> 4) << 3;
#pragma unroll
  for (int et = 0; et < 4; ++et)
#pragma unroll
    for (int r = 0; r < 8; ++r) {
      const int mrow = mblk * 128 + w * 16 + chi + r;
      attn[(size_t)(b * 8192 + mrow) * 512 + h * 64 + et * 16 + (lane & 15)] =
          acc[et].f[r];
    }
}

// ---------------------------------------------------------------------------
extern "C" void kernel_launch(void* const* d_in, const int* in_sizes, int n_in,
                              void* d_out, int out_size, void* d_ws, size_t ws_size,
                              hipStream_t stream) {
  (void)in_sizes; (void)n_in; (void)out_size; (void)ws_size;
  const float* x          = (const float*)d_in[0];
  const unsigned char* mk = (const unsigned char*)d_in[1];   // bool mask [B,H,N,1]
  const float* Wqkv       = (const float*)d_in[2];
  const float* Wout       = (const float*)d_in[3];
  const float* bout       = (const float*)d_in[4];
  float* out = (float*)d_out;

  float* qkv  = (float*)d_ws;                                // 32768*1536
  float* ctxp = qkv  + (size_t)32768 * 1536;                 // 8*32*4096
  float* ctx  = ctxp + (size_t)8 * 32 * 4096;                // 32*4096
  float* attn = ctx  + (size_t)32 * 4096;                    // 32768*512

  // 1) qkv = x @ Wqkv
  gemm_bf16_wmma<false><<<dim3(32768 / 128, 1536 / 128), 256, 0, stream>>>(
      x, Wqkv, nullptr, qkv, 32768, 1536, 512);
  // 2) q softmax over d (in place)
  q_softmax_kernel<<<32768, 256, 0, stream>>>(qkv);
  // 3) k softmax over n + mask (in place)
  k_softmax_kernel<<<2048, 256, 0, stream>>>(qkv, mk);
  // 4) ctx = q^T @ k  (deterministic split-K, fixed-order reduce)
  ctx_partial_kernel<<<dim3(32, 8), 256, 0, stream>>>(qkv, ctxp);
  ctx_reduce_kernel<<<131072 / 256, 256, 0, stream>>>(ctxp, ctx);
  // 5) att = v @ ctx
  attn_out_kernel<<<dim3(64, 32), 256, 0, stream>>>(qkv, ctx, attn);
  // 6) out = att @ Wout + bout
  gemm_bf16_wmma<true><<<dim3(32768 / 128, 512 / 128), 256, 0, stream>>>(
      attn, Wout, bout, out, 32768, 512, 512);
}
